// BiMambaEncoder_74818330296769
// MI455X (gfx1250) — matrix operand order repaired and checked by reading, not compiled, not run on hardware
//
#include <hip/hip_runtime.h>
#include <cstdint>
#include <cstddef>

// Problem constants (match reference)
#define B_   4
#define L_   4096
#define DM   512          // d_model
#define DI   1024         // d_inner
#define DS_  16           // d_state
#define DTR  32           // dt_rank
#define DCONV 4
#define MT   (B_*L_)      // 16384 rows for all GEMMs

typedef __attribute__((ext_vector_type(16))) __bf16 v16bf;
typedef __attribute__((ext_vector_type(8)))  float  v8f;
typedef __attribute__((ext_vector_type(4)))  int    v4i;

#define LDS_AS __attribute__((address_space(3)))
#define GLB_AS __attribute__((address_space(1)))

#if defined(__has_builtin)
#if __has_builtin(__builtin_amdgcn_global_load_async_to_lds_b128)
#define ASYNC_LDS 1
#endif
#endif

__device__ __forceinline__ unsigned short f2bf(float f) {
  union { float f; uint32_t u; } c; c.f = f;
  uint32_t u = c.u + 0x7FFFu + ((c.u >> 16) & 1u);   // round-to-nearest-even
  return (unsigned short)(u >> 16);
}

__device__ __forceinline__ float silu_f(float x) { return x / (1.0f + __expf(-x)); }

template<int EPI>
__device__ __forceinline__ float epi_f(float x) {
  if constexpr (EPI == 2) {          // exact GELU
    return 0.5f * x * (1.0f + erff(x * 0.70710678118654752f));
  } else if constexpr (EPI == 3) {   // softplus
    return (x > 20.0f) ? x : __logf(1.0f + __expf(x));
  }
  return x;
}

// 16-byte global(bf16) -> LDS copy; async-to-LDS when the toolchain has it.
__device__ __forceinline__ void cp16_g2l(const unsigned short* gsrc,
                                         unsigned short* ldst) {
#if defined(ASYNC_LDS)
  __builtin_amdgcn_global_load_async_to_lds_b128(
      (GLB_AS v4i*)(uintptr_t)gsrc,
      (LDS_AS v4i*)(uint32_t)(uintptr_t)ldst, 0, 0);
#else
  *(uint4*)ldst = *(const uint4*)gsrc;
#endif
}

// Wait until at most N async-to-LDS ops from this wave are outstanding.
template<int N>
__device__ __forceinline__ void async_wait() {
#if defined(ASYNC_LDS)
#if __has_builtin(__builtin_amdgcn_s_wait_asynccnt)
  __builtin_amdgcn_s_wait_asynccnt(N);
#else
  asm volatile("s_wait_asynccnt %0" :: "i"(N) : "memory");
#endif
#endif
}

// ---------------------------------------------------------------------------
// GEMM: C[M,N] = epi(Abf[M,lda(K cols)] * Wbf[N,K]^T + bias), bf16 in, f32 acc
// Requirements: M%128==0, N%64==0, K%32==0 (true for all call sites here).
// Block: 256 threads (8 waves). Tile: 128M x 64N, BK=32, LDS double-buffered.
// Wave grid: 4 along M x 2 along N; each wave owns 32x32 -> 4 WMMAs/K-step.
// Pipeline: issue async loads for tile k+1, wait ASYNCcnt<=3 (tile k done,
// 3 async ops/tile/wave, in-order completion), compute tile k from LDS while
// tile k+1 streams from HBM.
// ---------------------------------------------------------------------------
template<int EPI, bool OF32, bool OBF>
__global__ __launch_bounds__(256) void gemm_bf16_wmma(
    const unsigned short* __restrict__ A, int lda,
    const unsigned short* __restrict__ W,
    const float* __restrict__ bias,
    float* __restrict__ Cf,
    unsigned short* __restrict__ Cb,
    int M, int N, int K)
{
  __shared__ __align__(16) unsigned short sA[2][128][32];  // 2 x 8 KB
  __shared__ __align__(16) unsigned short sB[2][64][32];   // 2 x 4 KB

  const int tid   = threadIdx.x;
  const int lane  = tid & 31;
  const int wave  = tid >> 5;
  const int waveM = wave & 3;   // 0..3 (32 rows each)
  const int waveN = wave >> 2;  // 0..1 (32 cols each)

  const int m0 = blockIdx.y * 128;
  const int n0 = blockIdx.x * 64;

  // cooperative tile staging coords (16B chunks of 8 bf16)
  const int arow0 = tid >> 2;             // chunk tid   : rows 0..63
  const int arow1 = (tid + 256) >> 2;     // chunk tid+256: rows 64..127
  const int aseg  = (tid & 3) * 8;        // 0,8,16,24

  const int h  = lane >> 4;               // lane half (0/1)
  const int ml = lane & 15;

  v8f acc00 = {}, acc01 = {}, acc10 = {}, acc11 = {};

  auto issue_tile = [&](int k0, int buf) {
    const unsigned short* gA0 = A + (size_t)(m0 + arow0) * lda + (k0 + aseg);
    const unsigned short* gA1 = A + (size_t)(m0 + arow1) * lda + (k0 + aseg);
    const unsigned short* gW  = W + (size_t)(n0 + arow0) * K   + (k0 + aseg);
    cp16_g2l(gA0, &sA[buf][arow0][aseg]);
    cp16_g2l(gA1, &sA[buf][arow1][aseg]);
    cp16_g2l(gW,  &sB[buf][arow0][aseg]);
  };

  issue_tile(0, 0);

  for (int k0 = 0; k0 < K; k0 += 32) {
    const int buf = (k0 >> 5) & 1;
    if (k0 + 32 < K) {
      issue_tile(k0 + 32, buf ^ 1);     // prefetch next tile into other buffer
      async_wait<3>();                  // current tile's 3 ops complete
    } else {
      async_wait<0>();                  // last tile: drain
    }
    __syncthreads();                    // all waves' stores to LDS visible

    // A operand (16x32 bf16): lane(h,ml): 16B runs at k=8h and k=16+8h.
    union { v16bf v; uint4 q[2]; } a0, a1, b0, b1;
    const int ra = waveM * 32 + ml;
    a0.q[0] = *(const uint4*)&sA[buf][ra][8 * h];
    a0.q[1] = *(const uint4*)&sA[buf][ra][16 + 8 * h];
    a1.q[0] = *(const uint4*)&sA[buf][ra + 16][8 * h];
    a1.q[1] = *(const uint4*)&sA[buf][ra + 16][16 + 8 * h];

    // B operand (32x16 bf16): lane(h, n=ml): 16B runs at k=8h and k=16+8h.
    const int rb = waveN * 32 + ml;
    b0.q[0] = *(const uint4*)&sB[buf][rb][8 * h];
    b0.q[1] = *(const uint4*)&sB[buf][rb][16 + 8 * h];
    b1.q[0] = *(const uint4*)&sB[buf][rb + 16][8 * h];
    b1.q[1] = *(const uint4*)&sB[buf][rb + 16][16 + 8 * h];

    acc00 = __builtin_amdgcn_wmma_f32_16x16x32_bf16(false, a0.v, false, b0.v,
                                                    (short)0, acc00, false, false);
    acc01 = __builtin_amdgcn_wmma_f32_16x16x32_bf16(false, a0.v, false, b1.v,
                                                    (short)0, acc01, false, false);
    acc10 = __builtin_amdgcn_wmma_f32_16x16x32_bf16(false, a1.v, false, b0.v,
                                                    (short)0, acc10, false, false);
    acc11 = __builtin_amdgcn_wmma_f32_16x16x32_bf16(false, a1.v, false, b1.v,
                                                    (short)0, acc11, false, false);
    __syncthreads();   // all reads done before this buffer is refilled (k0+2)
  }

  // C/D layout: VGPR v -> row m = v + 8h ; lane ml carries n.
  const int gm = m0 + waveM * 32 + 8 * h;
  const int gn = n0 + waveN * 32 + ml;

  float bia0 = 0.0f, bia1 = 0.0f;
  if constexpr (EPI >= 1) { bia0 = bias[gn]; bia1 = bias[gn + 16]; }

  #pragma unroll
  for (int v = 0; v < 8; ++v) {
    const size_t r0 = (size_t)(gm + v) * N;
    const size_t r1 = (size_t)(gm + v + 16) * N;
    const float v00 = epi_f<EPI>(acc00[v] + bia0);
    const float v01 = epi_f<EPI>(acc01[v] + bia1);
    const float v10 = epi_f<EPI>(acc10[v] + bia0);
    const float v11 = epi_f<EPI>(acc11[v] + bia1);
    if constexpr (OF32) {
      Cf[r0 + gn] = v00; Cf[r0 + gn + 16] = v01;
      Cf[r1 + gn] = v10; Cf[r1 + gn + 16] = v11;
    }
    if constexpr (OBF) {
      Cb[r0 + gn] = f2bf(v00); Cb[r0 + gn + 16] = f2bf(v01);
      Cb[r1 + gn] = f2bf(v10); Cb[r1 + gn + 16] = f2bf(v11);
    }
  }
}

// ---------------------------------------------------------------------------
// fp32 -> bf16 elementwise (weights, one-shot)
// ---------------------------------------------------------------------------
__global__ __launch_bounds__(256) void cvt_bf16(
    const float* __restrict__ src, unsigned short* __restrict__ dst, int n)
{
  const int i = blockIdx.x * 256 + threadIdx.x;
  if (i < n) dst[i] = f2bf(src[i]);
}

// ---------------------------------------------------------------------------
// LayerNorm of x -> xn (bf16) and its time-flipped copy (bf16)
// ---------------------------------------------------------------------------
__global__ __launch_bounds__(256) void ln_fwd_rev(
    const float* __restrict__ x, const float* __restrict__ w,
    const float* __restrict__ b, unsigned short* __restrict__ xn,
    unsigned short* __restrict__ xnrev)
{
  __shared__ float red[256];
  const int row = blockIdx.x;                 // 0..MT-1
  const int bb = row / L_, l = row % L_;
  const float* xr = x + (size_t)row * DM;

  float s = 0.0f;
  for (int i = threadIdx.x; i < DM; i += 256) s += xr[i];
  red[threadIdx.x] = s; __syncthreads();
  for (int o = 128; o > 0; o >>= 1) {
    if (threadIdx.x < o) red[threadIdx.x] += red[threadIdx.x + o];
    __syncthreads();
  }
  const float mean = red[0] * (1.0f / DM);
  __syncthreads();

  float v = 0.0f;
  for (int i = threadIdx.x; i < DM; i += 256) { float d = xr[i] - mean; v += d * d; }
  red[threadIdx.x] = v; __syncthreads();
  for (int o = 128; o > 0; o >>= 1) {
    if (threadIdx.x < o) red[threadIdx.x] += red[threadIdx.x + o];
    __syncthreads();
  }
  const float rstd = rsqrtf(red[0] * (1.0f / DM) + 1e-5f);

  const size_t o1 = (size_t)row * DM;
  const size_t o2 = ((size_t)bb * L_ + (L_ - 1 - l)) * DM;
  for (int i = threadIdx.x; i < DM; i += 256) {
    const unsigned short y = f2bf((xr[i] - mean) * rstd * w[i] + b[i]);
    xn[o1 + i] = y;
    xnrev[o2 + i] = y;
  }
}

// ---------------------------------------------------------------------------
// Depthwise causal conv (K=4) + bias + SiLU -> xc (f32 for scan) + bf16 copy
// ---------------------------------------------------------------------------
__global__ __launch_bounds__(256) void conv_silu(
    const float* __restrict__ xin, const float* __restrict__ cw,
    const float* __restrict__ cb, float* __restrict__ xc,
    unsigned short* __restrict__ xcb)
{
  const size_t idx = (size_t)blockIdx.x * 256 + threadIdx.x;  // over B*L*DI
  if (idx >= (size_t)MT * DI) return;
  const int d = (int)(idx % DI);
  const int l = (int)((idx / DI) % L_);
  float acc = cb[d];
  #pragma unroll
  for (int k = 0; k < DCONV; ++k) {
    const int ls = l - (DCONV - 1) + k;
    const float xv = (ls >= 0) ? xin[idx + (size_t)(ls - l) * DI] : 0.0f;
    acc += xv * cw[d * DCONV + k];
  }
  const float r = silu_f(acc);
  xc[idx]  = r;
  xcb[idx] = f2bf(r);
}

// ---------------------------------------------------------------------------
// Selective scan fused with z-gating -> y (bf16, feeds out_proj GEMM only).
// grid = B_*4 blocks x 256 threads; thread owns channel d, 16 states in regs.
// B(t)/C(t) staged in LDS 8 timesteps per barrier.
// ---------------------------------------------------------------------------
__global__ __launch_bounds__(256) void scan_gate(
    const float* __restrict__ u,     // xc   (B,L,DI)
    const float* __restrict__ dt,    // (B,L,DI)
    const float* __restrict__ xdbl,  // (B,L,64): Bm cols 32..47, C cols 48..63
    const float* __restrict__ A_log, // (DI,16)
    const float* __restrict__ Dp,    // (DI)
    const float* __restrict__ z,     // (B,L,DI)
    unsigned short* __restrict__ y)  // (B,L,DI) bf16
{
  const int bb = blockIdx.x >> 2;
  const int d  = ((blockIdx.x & 3) << 8) + threadIdx.x;

  float Arow[DS_], h[DS_];
  #pragma unroll
  for (int n = 0; n < DS_; ++n) {
    Arow[n] = -__expf(A_log[d * DS_ + n]);
    h[n] = 0.0f;
  }
  const float Dd = Dp[d];

  __shared__ float sBC[8 * 32];
  for (int l0 = 0; l0 < L_; l0 += 8) {
    const size_t t0 = (size_t)bb * L_ + l0;
    // 256 threads fetch 8 steps x 32 (B|C) values
    sBC[threadIdx.x] = xdbl[(t0 + (threadIdx.x >> 5)) * 64 + 32 + (threadIdx.x & 31)];
    __syncthreads();

    #pragma unroll
    for (int s = 0; s < 8; ++s) {
      const size_t t = t0 + s;
      const float dtv = dt[t * DI + d];
      const float uv  = u[t * DI + d];
      const float* bc = &sBC[s * 32];
      float accy = 0.0f;
      #pragma unroll
      for (int n = 0; n < DS_; ++n) {
        const float dA = __expf(dtv * Arow[n]);
        h[n] = h[n] * dA + dtv * bc[n] * uv;
        accy += h[n] * bc[16 + n];
      }
      y[t * DI + d] = f2bf((accy + uv * Dd) * silu_f(z[t * DI + d]));
    }
    __syncthreads();
  }
}

// ---------------------------------------------------------------------------
// m = fwd + flip_L(bwd)  -> bf16 (feeds ff1 GEMM only)
// ---------------------------------------------------------------------------
__global__ __launch_bounds__(256) void add_flip(
    const float* __restrict__ f, const float* __restrict__ bwd,
    unsigned short* __restrict__ m)
{
  const size_t idx = (size_t)blockIdx.x * 256 + threadIdx.x;  // over B*L*DM
  if (idx >= (size_t)MT * DM) return;
  const int c = (int)(idx % DM);
  const size_t bl = idx / DM;
  const int l = (int)(bl % L_);
  const int bb = (int)(bl / L_);
  m[idx] = f2bf(f[idx] + bwd[((size_t)bb * L_ + (L_ - 1 - l)) * DM + c]);
}

// ---------------------------------------------------------------------------
// out = layernorm(ff)*w+b + residual x
// ---------------------------------------------------------------------------
__global__ __launch_bounds__(256) void ln_residual(
    const float* __restrict__ ff, const float* __restrict__ w,
    const float* __restrict__ b, const float* __restrict__ x,
    float* __restrict__ out)
{
  __shared__ float red[256];
  const int row = blockIdx.x;
  const float* fr = ff + (size_t)row * DM;

  float s = 0.0f;
  for (int i = threadIdx.x; i < DM; i += 256) s += fr[i];
  red[threadIdx.x] = s; __syncthreads();
  for (int o = 128; o > 0; o >>= 1) {
    if (threadIdx.x < o) red[threadIdx.x] += red[threadIdx.x + o];
    __syncthreads();
  }
  const float mean = red[0] * (1.0f / DM);
  __syncthreads();

  float v = 0.0f;
  for (int i = threadIdx.x; i < DM; i += 256) { float d = fr[i] - mean; v += d * d; }
  red[threadIdx.x] = v; __syncthreads();
  for (int o = 128; o > 0; o >>= 1) {
    if (threadIdx.x < o) red[threadIdx.x] += red[threadIdx.x + o];
    __syncthreads();
  }
  const float rstd = rsqrtf(red[0] * (1.0f / DM) + 1e-5f);

  const size_t o1 = (size_t)row * DM;
  for (int i = threadIdx.x; i < DM; i += 256)
    out[o1 + i] = (fr[i] - mean) * rstd * w[i] + b[i] + x[o1 + i];
}

// ---------------------------------------------------------------------------
// Host-side orchestration
// ---------------------------------------------------------------------------
extern "C" void kernel_launch(void* const* d_in, const int* in_sizes, int n_in,
                              void* d_out, int out_size, void* d_ws, size_t ws_size,
                              hipStream_t stream) {
  const float* x         = (const float*)d_in[0];
  const float* in_proj_w = (const float*)d_in[1];   // (2048, 512)
  const float* conv_w    = (const float*)d_in[2];   // (1024, 4)
  const float* conv_b    = (const float*)d_in[3];
  const float* x_proj_w  = (const float*)d_in[4];   // (64, 1024)
  const float* dt_proj_w = (const float*)d_in[5];   // (1024, 32)
  const float* dt_proj_b = (const float*)d_in[6];
  const float* A_log     = (const float*)d_in[7];   // (1024, 16)
  const float* Dp        = (const float*)d_in[8];
  const float* out_proj_w= (const float*)d_in[9];   // (512, 1024)
  const float* ln1_w     = (const float*)d_in[10];
  const float* ln1_b     = (const float*)d_in[11];
  const float* ln2_w     = (const float*)d_in[12];
  const float* ln2_b     = (const float*)d_in[13];
  const float* ff_w1     = (const float*)d_in[14];  // (2048, 512)
  const float* ff_b1     = (const float*)d_in[15];
  const float* ff_w2     = (const float*)d_in[16];  // (512, 2048)
  const float* ff_b2     = (const float*)d_in[17];
  float* out = (float*)d_out;
  (void)ws_size; (void)n_in; (void)in_sizes; (void)out_size;

  const size_t SZ_DM = (size_t)MT * DM;    // 8,388,608 elems
  const size_t SZ_DI = (size_t)MT * DI;    // 16,777,216 elems
  const size_t SZ_XD = (size_t)MT * 64;    // 1,048,576 elems

  // Arena: carve fp32 then bf16 regions (all 16B aligned by construction).
  char* p = (char*)d_ws;
  auto carve = [&](size_t bytes) { char* r = p; p += (bytes + 63) & ~(size_t)63; return r; };

  float* xin  = (float*)carve(SZ_DI * 4);
  float* zb   = (float*)carve(SZ_DI * 4);
  float* xc   = (float*)carve(SZ_DI * 4);
  float* xdbl = (float*)carve(SZ_XD * 4);
  float* dtb  = (float*)carve(SZ_DI * 4);
  float* out2 = (float*)carve(2 * SZ_DM * 4);
  float* ffb  = (float*)carve(SZ_DM * 4);

  unsigned short* xn_bf    = (unsigned short*)carve(SZ_DM * 2);
  unsigned short* xnrev_bf = (unsigned short*)carve(SZ_DM * 2);
  unsigned short* xc_bf    = (unsigned short*)carve(SZ_DI * 2);
  unsigned short* xdbl_bf  = (unsigned short*)carve(SZ_XD * 2);
  unsigned short* y_bf     = (unsigned short*)carve(SZ_DI * 2);
  unsigned short* mb_bf    = (unsigned short*)carve(SZ_DM * 2);
  unsigned short* hb_bf    = (unsigned short*)carve((size_t)MT * 2048 * 2);

  unsigned short* w_in  = (unsigned short*)carve((size_t)2 * DI * DM * 2);
  unsigned short* w_xp  = (unsigned short*)carve((size_t)64 * DI * 2);
  unsigned short* w_dt  = (unsigned short*)carve((size_t)DI * DTR * 2);
  unsigned short* w_op  = (unsigned short*)carve((size_t)DM * DI * 2);
  unsigned short* w_f1  = (unsigned short*)carve((size_t)2048 * DM * 2);
  unsigned short* w_f2  = (unsigned short*)carve((size_t)DM * 2048 * 2);

  // 0) one-shot weight conversion to bf16
  auto cvt = [&](const float* s, unsigned short* d, int n) {
    cvt_bf16<<<(n + 255) / 256, 256, 0, stream>>>(s, d, n);
  };
  cvt(in_proj_w, w_in, 2 * DI * DM);
  cvt(x_proj_w,  w_xp, 64 * DI);
  cvt(dt_proj_w, w_dt, DI * DTR);
  cvt(out_proj_w,w_op, DM * DI);
  cvt(ff_w1,     w_f1, 2048 * DM);
  cvt(ff_w2,     w_f2, DM * 2048);

  // 1) LayerNorm + flipped copy (bf16 outputs feed GEMMs only)
  ln_fwd_rev<<<MT, 256, 0, stream>>>(x, ln1_w, ln1_b, xn_bf, xnrev_bf);

  // Per-direction Mamba
  for (int dir = 0; dir < 2; ++dir) {
    const unsigned short* src = (dir == 0) ? xn_bf : xnrev_bf;
    float* dir_out = out2 + (size_t)dir * SZ_DM;

    // in_proj split into x-part and z-part (weights are (2*DI, DM))
    gemm_bf16_wmma<0, true, false><<<dim3(DI / 64, MT / 128), 256, 0, stream>>>(
        src, DM, w_in, nullptr, xin, nullptr, MT, DI, DM);
    gemm_bf16_wmma<0, true, false><<<dim3(DI / 64, MT / 128), 256, 0, stream>>>(
        src, DM, w_in + (size_t)DI * DM, nullptr, zb, nullptr, MT, DI, DM);

    // conv + SiLU (f32 for scan u, bf16 for x_proj GEMM)
    conv_silu<<<(unsigned)((SZ_DI + 255) / 256), 256, 0, stream>>>(
        xin, conv_w, conv_b, xc, xc_bf);

    // x_proj: (M,1024) x (64,1024)^T -> (M,64): f32 (scan B/C) + bf16 (dt GEMM)
    gemm_bf16_wmma<0, true, true><<<dim3(64 / 64, MT / 128), 256, 0, stream>>>(
        xc_bf, DI, w_xp, nullptr, xdbl, xdbl_bf, MT, 64, DI);

    // dt = softplus(dt_r @ dt_proj_w^T + b);  dt_r = xdbl cols [0,32), lda=64
    gemm_bf16_wmma<3, true, false><<<dim3(DI / 64, MT / 128), 256, 0, stream>>>(
        xdbl_bf, 64, w_dt, dt_proj_b, dtb, nullptr, MT, DI, DTR);

    // selective scan + u*Dp + silu(z) gating -> y (bf16)
    scan_gate<<<B_ * 4, 256, 0, stream>>>(xc, dtb, xdbl, A_log, Dp, zb, y_bf);

    // out_proj: (M,1024) x (512,1024)^T -> (M,512) f32
    gemm_bf16_wmma<0, true, false><<<dim3(DM / 64, MT / 128), 256, 0, stream>>>(
        y_bf, DI, w_op, nullptr, dir_out, nullptr, MT, DM, DI);
  }

  // m = fwd + flip(bwd) -> bf16
  add_flip<<<(unsigned)((SZ_DM + 255) / 256), 256, 0, stream>>>(
      out2, out2 + SZ_DM, mb_bf);

  // FFN: h = gelu(m @ ff_w1^T + b1) -> bf16 ; ff = h @ ff_w2^T + b2 -> f32
  gemm_bf16_wmma<2, false, true><<<dim3(2048 / 64, MT / 128), 256, 0, stream>>>(
      mb_bf, DM, w_f1, ff_b1, nullptr, hb_bf, MT, 2048, DM);
  gemm_bf16_wmma<1, true, false><<<dim3(DM / 64, MT / 128), 256, 0, stream>>>(
      hb_bf, 2048, w_f2, ff_b2, ffb, nullptr, MT, DM, 2048);

  // final LN + residual
  ln_residual<<<MT, 256, 0, stream>>>(ffb, ln2_w, ln2_b, x, out);
}
